// GraphTransformerLayer_35407710388433
// MI455X (gfx1250) — matrix operand order repaired and checked by reading, not compiled, not run on hardware
//
#include <hip/hip_runtime.h>
#include <hip/hip_bf16.h>

typedef __attribute__((ext_vector_type(16))) __bf16 v16bf;
typedef __attribute__((ext_vector_type(8)))  __bf16 v8bf;
typedef __attribute__((ext_vector_type(8)))  float  v8f;

__device__ inline void atomicAddF(float* p, float v) {
    __hip_atomic_fetch_add(p, v, __ATOMIC_RELAXED, __HIP_MEMORY_SCOPE_AGENT);
}

__device__ inline float4 ld4(const float* p) {
    return *reinterpret_cast<const float4*>(p);
}

union Frag16 { v16bf v; v8bf h[2]; };

// ---------------------------------------------------------------------------
// Elementwise f32 -> (bf16 hi, bf16 lo) split:  x ~= hi + lo, |err| ~ 2^-16 x
// ---------------------------------------------------------------------------
__global__ __launch_bounds__(256) void split_f32(
    const float* __restrict__ X, __bf16* __restrict__ Xh, __bf16* __restrict__ Xl, size_t n)
{
    size_t i = (size_t)blockIdx.x * blockDim.x + threadIdx.x;
    if (i >= n) return;
    float x = X[i];
    __bf16 hb = (__bf16)x;
    Xh[i] = hb;
    Xl[i] = (__bf16)(x - (float)hb);
}

// ---------------------------------------------------------------------------
// Register-strip GEMM: Out[M,OC] = act(A[M,KC] @ W[OC,KC]^T + bias) (+ resid)
// A and W pre-split into bf16 hi/lo; 3 WMMAs per 32-wide K step recover ~f32.
// One wave computes a full 16xOC row strip: A fragments loaded once per K
// step and reused across all OC/16 column tiles (accumulators in registers).
// ---------------------------------------------------------------------------
template<int KC, int OC, bool RELU, bool SPLITOUT>
__global__ __launch_bounds__(256) void gemm_rs(
    const __bf16* __restrict__ Ah, const __bf16* __restrict__ Al,
    const __bf16* __restrict__ Wh, const __bf16* __restrict__ Wl,
    const float* __restrict__ bias, const float* __restrict__ resid,
    float* __restrict__ Outf, __bf16* __restrict__ Outh, __bf16* __restrict__ Outl,
    int Mtiles)
{
    constexpr int OT = OC / 16;
    const int widx = threadIdx.x >> 5;
    const int lane = threadIdx.x & 31;
    const int trow = blockIdx.x * 8 + widx;
    if (trow >= Mtiles) return;

    const int m  = lane & 15;   // A row within tile / B column within tile
    const int hi = lane >> 4;   // lane-half selects K sub-blocks

    const __bf16* arh = Ah + (size_t)(trow * 16 + m) * KC;
    const __bf16* arl = Al + (size_t)(trow * 16 + m) * KC;

    v8f zero = {};
    v8f acc[OT];
#pragma unroll
    for (int t = 0; t < OT; ++t) acc[t] = zero;

#pragma unroll
    for (int ks = 0; ks < KC / 32; ++ks) {
        const int ka = ks * 32 + hi * 16;   // A: K = ka..ka+15
        const int kb = ks * 32 + hi * 16;   // B: K = kb..kb+15

        Frag16 fah, fal;
        fah.h[0] = *(const v8bf*)(arh + ka);
        fah.h[1] = *(const v8bf*)(arh + ka + 8);
        fal.h[0] = *(const v8bf*)(arl + ka);
        fal.h[1] = *(const v8bf*)(arl + ka + 8);
        v16bf ah = fah.v, al = fal.v;

#pragma unroll
        for (int t = 0; t < OT; ++t) {
            const __bf16* wrh = Wh + (size_t)(t * 16 + m) * KC + kb;
            const __bf16* wrl = Wl + (size_t)(t * 16 + m) * KC + kb;
            v16bf bh = *(const v16bf*)wrh;
            v16bf bl = *(const v16bf*)wrl;
            acc[t] = __builtin_amdgcn_wmma_f32_16x16x32_bf16(false, ah, false, bh, (short)0, acc[t], false, false);
            acc[t] = __builtin_amdgcn_wmma_f32_16x16x32_bf16(false, ah, false, bl, (short)0, acc[t], false, false);
            acc[t] = __builtin_amdgcn_wmma_f32_16x16x32_bf16(false, al, false, bh, (short)0, acc[t], false, false);
        }
    }

#pragma unroll
    for (int t = 0; t < OT; ++t) {
        const int col = t * 16 + m;
        const float bb = bias ? bias[col] : 0.0f;
#pragma unroll
        for (int r = 0; r < 8; ++r) {
            const int row = trow * 16 + r + hi * 8;
            float v = acc[t][r] + bb;
            if (resid) v += resid[(size_t)row * OC + col];
            if (RELU)  v = fmaxf(v, 0.0f);
            if (Outf)  Outf[(size_t)row * OC + col] = v;
            if (SPLITOUT) {
                __bf16 vh = (__bf16)v;
                Outh[(size_t)row * OC + col] = vh;
                Outl[(size_t)row * OC + col] = (__bf16)(v - (float)vh);
            }
        }
    }
}

// ---------------------------------------------------------------------------
// Fused edge kernel: per (edge, head) score = exp(clip(K[src].Q[dst]/4)),
// scatter V[src]*score into wV[dst] and score into z[dst] (L2 atomics).
// ---------------------------------------------------------------------------
__global__ __launch_bounds__(256) void edge_attn(
    const float* __restrict__ Q, const float* __restrict__ K,
    const float* __restrict__ V, const int* __restrict__ src,
    const int* __restrict__ dst, float* __restrict__ wV,
    float* __restrict__ z, int E)
{
    size_t t = (size_t)blockIdx.x * blockDim.x + threadIdx.x;
    if (t >= (size_t)E * 8) return;
    const int e  = (int)(t >> 3);
    const int hd = (int)(t & 7);
    const int s  = src[e];
    const int d  = dst[e];

    const float* kp = K + (size_t)s * 128 + hd * 16;
    const float* qp = Q + (size_t)d * 128 + hd * 16;
    float acc = 0.0f;
#pragma unroll
    for (int j = 0; j < 4; ++j) {
        float4 a = ld4(kp + 4 * j);
        float4 b = ld4(qp + 4 * j);
        acc += a.x * b.x + a.y * b.y + a.z * b.z + a.w * b.w;
    }
    float sc = acc * 0.25f;                       // 1/sqrt(16)
    sc = fminf(5.0f, fmaxf(-5.0f, sc));
    sc = expf(sc);

    atomicAddF(z + (size_t)d * 8 + hd, sc);

    const float* vp = V + (size_t)s * 128 + hd * 16;
    float* wp = wV + (size_t)d * 128 + hd * 16;
#pragma unroll
    for (int j = 0; j < 4; ++j) {
        float4 v = ld4(vp + 4 * j);
        atomicAddF(wp + 4 * j + 0, v.x * sc);
        atomicAddF(wp + 4 * j + 1, v.y * sc);
        atomicAddF(wp + 4 * j + 2, v.z * sc);
        atomicAddF(wp + 4 * j + 3, v.w * sc);
    }
}

// attn = wV / z, emitted directly as bf16 hi/lo fragments for the O-proj GEMM
__global__ __launch_bounds__(256) void attn_div_split(
    const float* __restrict__ wV, const float* __restrict__ z,
    __bf16* __restrict__ Ah, __bf16* __restrict__ Al, size_t n)
{
    size_t i = (size_t)blockIdx.x * blockDim.x + threadIdx.x;
    if (i >= n) return;
    size_t node = i >> 7;
    int hd = (int)((i >> 4) & 7);
    float v = wV[i] / z[node * 8 + hd];
    __bf16 vh = (__bf16)v;
    Ah[i] = vh;
    Al[i] = (__bf16)(v - (float)vh);
}

// per-column sum / sum-of-squares partial reduction (C = 128)
__global__ __launch_bounds__(128) void colstats(
    const float* __restrict__ X, int M, float* __restrict__ stats)
{
    const int c  = threadIdx.x;
    const int r0 = blockIdx.x * 250;
    const int r1 = min(r0 + 250, M);
    float s = 0.0f, s2 = 0.0f;
    for (int r = r0; r < r1; ++r) {
        float v = X[(size_t)r * 128 + c];
        s += v;
        s2 += v * v;
    }
    atomicAddF(stats + c, s);
    atomicAddF(stats + 128 + c, s2);
}

// stats -> per-column scale/shift: y = x*params[c] + params[128+c]
__global__ __launch_bounds__(128) void bnparams(
    const float* __restrict__ stats, const float* __restrict__ gam,
    const float* __restrict__ bet, float Nf, float* __restrict__ params)
{
    const int c = threadIdx.x;
    float mu  = stats[c] / Nf;
    float var = stats[128 + c] / Nf - mu * mu;
    float sc  = gam[c] * rsqrtf(var + 1e-5f);
    params[c] = sc;
    params[128 + c] = bet[c] - mu * sc;
}

// BN apply; optionally also emits bf16 hi/lo split of the result
__global__ __launch_bounds__(256) void bn_apply_split(
    const float* __restrict__ X, const float* __restrict__ params,
    float* __restrict__ Yf, __bf16* __restrict__ Yh, __bf16* __restrict__ Yl, size_t n)
{
    size_t i = (size_t)blockIdx.x * blockDim.x + threadIdx.x;
    if (i >= n) return;
    int c = (int)(i & 127);
    float v = X[i] * params[c] + params[128 + c];
    Yf[i] = v;
    if (Yh) {
        __bf16 vh = (__bf16)v;
        Yh[i] = vh;
        Yl[i] = (__bf16)(v - (float)vh);
    }
}

// ---------------------------------------------------------------------------
extern "C" void kernel_launch(void* const* d_in, const int* in_sizes, int n_in,
                              void* d_out, int out_size, void* d_ws, size_t ws_size,
                              hipStream_t stream)
{
    const float* h   = (const float*)d_in[0];
    const int*   src = (const int*)  d_in[1];
    const int*   dst = (const int*)  d_in[2];
    const float* WQ  = (const float*)d_in[3];
    const float* WK  = (const float*)d_in[4];
    const float* WV  = (const float*)d_in[5];
    const float* WO  = (const float*)d_in[6];
    const float* bO  = (const float*)d_in[7];
    const float* W1  = (const float*)d_in[8];
    const float* b1  = (const float*)d_in[9];
    const float* W2  = (const float*)d_in[10];
    const float* b2  = (const float*)d_in[11];
    const float* g1  = (const float*)d_in[12];
    const float* be1 = (const float*)d_in[13];
    const float* g2  = (const float*)d_in[14];
    const float* be2 = (const float*)d_in[15];

    const int N = in_sizes[0] / 128;    // 50000
    const int E = in_sizes[1];          // 640000
    const size_t ND = (size_t)N * 128;

    // ---- workspace layout (time-aliased) ----
    float* ws = (float*)d_ws;
    float* R0 = ws;                     // ND: Qb -> attn hi/lo -> out_pre
    float* R1 = R0 + ND;                // ND: Kb -> h1 hi/lo
    float* R2 = R1 + ND;                // ND: Vb -> g hi (N*256 bf16)
    float* R3 = R2 + ND;                // ND: wV -> h1 f32
    float* zb = R3 + ND;                // N*8
    float* R5 = zb + (size_t)N * 8;     // ND: t1 -> g lo (N*256 bf16)
    float* R6 = R5 + ND;                // ND: h hi/lo (bf16)
    float* Wsp = R6 + ND;               // weight splits: 262144 bf16 = 131072 f32
    float* stats  = Wsp + 131072;       // 256
    float* params = stats + 256;        // 256

    float* Qb = R0;  float* Kb = R1;  float* Vb = R2;
    float* wV = R3;  float* t1 = R5;
    __bf16* attnh = (__bf16*)R0;  __bf16* attnl = attnh + ND;
    __bf16* h1h   = (__bf16*)R1;  __bf16* h1l   = h1h + ND;
    float*  h1f   = R3;
    __bf16* gh    = (__bf16*)R2;                       // N*256 bf16
    __bf16* gl    = (__bf16*)R5;                       // N*256 bf16
    float*  outp  = R0;
    __bf16* hh    = (__bf16*)R6;  __bf16* hl = hh + ND;

    __bf16* WQh = (__bf16*)Wsp;        __bf16* WQl = WQh + 16384;
    __bf16* WKh = WQl + 16384;         __bf16* WKl = WKh + 16384;
    __bf16* WVh = WKl + 16384;         __bf16* WVl = WVh + 16384;
    __bf16* WOh = WVl + 16384;         __bf16* WOl = WOh + 16384;
    __bf16* W1h = WOl + 16384;         __bf16* W1l = W1h + 32768;
    __bf16* W2h = W1l + 32768;         __bf16* W2l = W2h + 32768;

    const int Mtiles = N / 16;          // 3125
    const int GB     = (Mtiles + 7) / 8;
    const int EWB    = (int)((ND + 255) / 256);

    // ---- pre-split inputs and weights to bf16 hi/lo ----
    split_f32<<<EWB, 256, 0, stream>>>(h, hh, hl, ND);
    split_f32<<<64, 256, 0, stream>>>(WQ, WQh, WQl, 16384);
    split_f32<<<64, 256, 0, stream>>>(WK, WKh, WKl, 16384);
    split_f32<<<64, 256, 0, stream>>>(WV, WVh, WVl, 16384);
    split_f32<<<64, 256, 0, stream>>>(WO, WOh, WOl, 16384);
    split_f32<<<128, 256, 0, stream>>>(W1, W1h, W1l, 32768);
    split_f32<<<128, 256, 0, stream>>>(W2, W2h, W2l, 32768);

    // ---- Q/K/V projections ----
    gemm_rs<128, 128, false, false><<<GB, 256, 0, stream>>>(hh, hl, WQh, WQl, nullptr, nullptr, Qb, nullptr, nullptr, Mtiles);
    gemm_rs<128, 128, false, false><<<GB, 256, 0, stream>>>(hh, hl, WKh, WKl, nullptr, nullptr, Kb, nullptr, nullptr, Mtiles);
    gemm_rs<128, 128, false, false><<<GB, 256, 0, stream>>>(hh, hl, WVh, WVl, nullptr, nullptr, Vb, nullptr, nullptr, Mtiles);

    // ---- edge attention + scatter ----
    (void)hipMemsetAsync(wV, 0, ND * sizeof(float), stream);
    (void)hipMemsetAsync(zb, 0, (size_t)N * 8 * sizeof(float), stream);
    const size_t eh = (size_t)E * 8;
    edge_attn<<<(int)((eh + 255) / 256), 256, 0, stream>>>(Qb, Kb, Vb, src, dst, wV, zb, E);
    attn_div_split<<<EWB, 256, 0, stream>>>(wV, zb, attnh, attnl, ND);

    // ---- O projection + residual -> t1; BN1 -> h1 (f32 + split) ----
    gemm_rs<128, 128, false, false><<<GB, 256, 0, stream>>>(attnh, attnl, WOh, WOl, bO, h, t1, nullptr, nullptr, Mtiles);
    (void)hipMemsetAsync(stats, 0, 256 * sizeof(float), stream);
    colstats<<<(N + 249) / 250, 128, 0, stream>>>(t1, N, stats);
    bnparams<<<1, 128, 0, stream>>>(stats, g1, be1, (float)N, params);
    bn_apply_split<<<EWB, 256, 0, stream>>>(t1, params, h1f, h1h, h1l, ND);

    // ---- FFN: g = relu(h1@W1^T + b1) (split-only out); out_pre = h1 + g@W2^T + b2 ----
    gemm_rs<128, 256, true,  true ><<<GB, 256, 0, stream>>>(h1h, h1l, W1h, W1l, b1, nullptr, nullptr, gh, gl, Mtiles);
    gemm_rs<256, 128, false, false><<<GB, 256, 0, stream>>>(gh, gl, W2h, W2l, b2, h1f, outp, nullptr, nullptr, Mtiles);

    // ---- BN2 -> d_out ----
    (void)hipMemsetAsync(stats, 0, 256 * sizeof(float), stream);
    colstats<<<(N + 249) / 250, 128, 0, stream>>>(outp, N, stats);
    bnparams<<<1, 128, 0, stream>>>(stats, g2, be2, (float)N, params);
    bn_apply_split<<<EWB, 256, 0, stream>>>(outp, params, (float*)d_out, nullptr, nullptr, ND);
}
